// CortexLM_76124000354324
// MI455X (gfx1250) — matrix operand order repaired and checked by reading, not compiled, not run on hardware
//
#include <hip/hip_runtime.h>

// ---------------------------------------------------------------------------
// BitNet-RWKV forward for MI455X (gfx1250, wave32, WMMA).
// Ternary weights are exact in bf16; GEMMs use v_wmma_f32_16x16x32_bf16 with
// double-buffered LDS, register-staged global loads, and sched_barrier-pinned
// fragment-load / WMMA groups (2 x 4 back-to-back WMMAs per K-stage).
// ---------------------------------------------------------------------------

typedef __bf16 bf16_t;
typedef bf16_t v16bf __attribute__((ext_vector_type(16)));
typedef float  v8f   __attribute__((ext_vector_type(8)));

#define GEMM_KS 72  // LDS row stride in bf16 elements: 144 bytes (16B aligned, bank-friendly)

static __device__ __forceinline__ unsigned short f2bf(float f) {
  union { float f; unsigned int u; } c; c.f = f;
  unsigned int u = c.u;
  if ((u & 0x7F800000u) == 0x7F800000u) return (unsigned short)(u >> 16); // inf/nan passthrough
  unsigned int r = u + 0x7FFFu + ((u >> 16) & 1u);                        // round-nearest-even
  return (unsigned short)(r >> 16);
}

static __device__ __forceinline__ float sigmoidf_(float x) {
  return 1.0f / (1.0f + __expf(-x));
}

// ---------------- elementwise fp32 -> bf16 ----------------
__global__ void cast_bf16_kernel(const float* __restrict__ x,
                                 unsigned short* __restrict__ y, long long n) {
  long long i = (long long)blockIdx.x * blockDim.x + threadIdx.x;
  long long s = (long long)gridDim.x * blockDim.x;
  for (; i < n; i += s) y[i] = f2bf(x[i]);
}

// ---------------- ternary row quantization -----------------
// s = max(mean|W_row|, 1e-5); w_q = clip(round(W/s), -1, 1) stored as bf16 (exact).
__global__ __launch_bounds__(256) void quantize_rows_kernel(
    const float* __restrict__ W, unsigned short* __restrict__ Wq,
    float* __restrict__ s, int K) {
  __shared__ float red[256];
  int row = blockIdx.x;
  const float* w = W + (size_t)row * K;
  float acc = 0.f;
  for (int k = threadIdx.x; k < K; k += 256) acc += fabsf(w[k]);
  red[threadIdx.x] = acc; __syncthreads();
  for (int off = 128; off > 0; off >>= 1) {
    if ((int)threadIdx.x < off) red[threadIdx.x] += red[threadIdx.x + off];
    __syncthreads();
  }
  float sc  = fmaxf(red[0] / (float)K, 1e-5f);
  float inv = 1.0f / sc;
  unsigned short* out = Wq + (size_t)row * K;
  for (int k = threadIdx.x; k < K; k += 256) {
    float q = fminf(fmaxf(rintf(w[k] * inv), -1.f), 1.f);
    out[k] = f2bf(q);
  }
  if (threadIdx.x == 0) s[row] = sc;
}

// ---------------- embed gather + rmsnorm (fp32 out) ----------------
__global__ __launch_bounds__(256) void gather_rmsnorm_kernel(
    const int* __restrict__ idx, const float* __restrict__ embed,
    const float* __restrict__ g, float* __restrict__ h, int D) {
  __shared__ float red[256];
  int m = blockIdx.x;
  const float* e = embed + (size_t)idx[m] * D;
  float acc = 0.f;
  for (int k = threadIdx.x; k < D; k += 256) { float v = e[k]; acc += v * v; }
  red[threadIdx.x] = acc; __syncthreads();
  for (int off = 128; off > 0; off >>= 1) {
    if ((int)threadIdx.x < off) red[threadIdx.x] += red[threadIdx.x + off];
    __syncthreads();
  }
  float inv = rsqrtf(red[0] / (float)D + 1e-6f);
  float* o = h + (size_t)m * D;
  for (int k = threadIdx.x; k < D; k += 256) o[k] = e[k] * inv * g[k];
}

// ---------------- rmsnorm -> bf16 (feeds WMMA A operand) ----------------
__global__ __launch_bounds__(256) void rmsnorm_cast_kernel(
    const float* __restrict__ x, const float* __restrict__ g,
    unsigned short* __restrict__ y, int D) {
  __shared__ float red[256];
  int m = blockIdx.x;
  const float* xr = x + (size_t)m * D;
  float acc = 0.f;
  for (int k = threadIdx.x; k < D; k += 256) { float v = xr[k]; acc += v * v; }
  red[threadIdx.x] = acc; __syncthreads();
  for (int off = 128; off > 0; off >>= 1) {
    if ((int)threadIdx.x < off) red[threadIdx.x] += red[threadIdx.x + off];
    __syncthreads();
  }
  float inv = rsqrtf(red[0] / (float)D + 1e-6f);
  unsigned short* o = y + (size_t)m * D;
  for (int k = threadIdx.x; k < D; k += 256) o[k] = f2bf(xr[k] * inv * g[k]);
}

// ---------------- WKV decay scan: rs[b,t,d] = r * (sum_tau dec^(t-tau) k*v) ----------------
__global__ void wkv_scan_kernel(const float* __restrict__ r, const float* __restrict__ k,
                                const float* __restrict__ v, const float* __restrict__ decay,
                                float* __restrict__ rs, int B, int T, int D) {
  int i = blockIdx.x * blockDim.x + threadIdx.x;
  if (i >= B * D) return;
  int d = i % D, b = i / D;
  float dec = sigmoidf_(decay[d]);
  size_t base = (size_t)b * T * D + d;
  float st = 0.f;
  for (int t = 0; t < T; ++t) {
    size_t o = base + (size_t)t * D;
    st = st * dec + k[o] * v[o];
    rs[o] = r[o] * st;
  }
}

// ---------------- tiled bf16 WMMA GEMM: C[M,N] = epi( A[M,K] * B[N,K]^T ) ----------------
// Block: 256 threads = 8 wave32s, tile 64(M) x 128(N), K-stage 64 (2 sub-slabs of 32).
// Double-buffered LDS, one barrier per K-stage. Each sub-slab: 10 ds_loads pinned
// before its 4 WMMAs with sched_barrier(0). Staging split: sa0/sa1/sb0/sb1 load in
// the gap between WMMA groups; sb2/sb3 load at the stage tail where fragments are
// dead (keeps peak pressure under the VGPR budget -> no scratch spills).
// Epilogue: v = acc * scale[n]; mode 1 -> sigmoid, mode 2 -> SiLU; then *mul, +add.
__global__ __launch_bounds__(256)
__attribute__((amdgpu_waves_per_eu(1)))
void gemm_bf16_kernel(
    const unsigned short* __restrict__ A, const unsigned short* __restrict__ Bw,
    const float* __restrict__ scale, const float* __restrict__ addp,
    const float* __restrict__ mulp, float* __restrict__ C,
    int M, int N, int K, int mode) {
  __shared__ __align__(16) unsigned short Alds[2][64 * GEMM_KS];    // 64 x 64 bf16 tiles
  __shared__ __align__(16) unsigned short Blds[2][128 * GEMM_KS];   // 128 x 64 bf16 tiles

  const int tid  = threadIdx.x;
  const int w    = tid >> 5;
  const int lane = tid & 31;
  const int wm   = w & 3;
  const int wn   = w >> 2;
  const int lr   = lane & 15;
  const int half = lane >> 4;
  const int bm   = blockIdx.y * 64;
  const int bn   = blockIdx.x * 128;
  if (bm >= M) return;

  v8f acc[4] = {};

  // Global staging: 16B chunks. A tile 64x64 = 512 chunks (2/thread),
  // B tile 128x64 = 1024 chunks (4/thread). Chunk c -> row c>>3, col (c&7)*8.
  const int ar0 = tid >> 3,          ac0 = (tid & 7) * 8;
  const int ar1 = (tid + 256) >> 3,  ac1 = ((tid + 256) & 7) * 8;
  const int lA0 = ar0 * GEMM_KS + ac0;
  const int lA1 = ar1 * GEMM_KS + ac1;
  // Running global pointers (advance by 64 elements per K-stage).
  const unsigned short* pa0 = A + (size_t)(bm + ar0) * K + ac0;
  const unsigned short* pa1 = A + (size_t)(bm + ar1) * K + ac1;
  const unsigned short* pb[4];
  int lB[4];
#pragma unroll
  for (int j = 0; j < 4; ++j) {
    const int c  = tid + j * 256;
    const int br = c >> 3;
    const int bc = (c & 7) * 8;
    pb[j] = Bw + (size_t)(bn + br) * K + bc;
    lB[j] = br * GEMM_KS + bc;
  }

  // Prologue: stage K-stage 0 into LDS buffer 0.
  {
    uint4 sa0 = *reinterpret_cast<const uint4*>(pa0);
    uint4 sa1 = *reinterpret_cast<const uint4*>(pa1);
    uint4 sb0 = *reinterpret_cast<const uint4*>(pb[0]);
    uint4 sb1 = *reinterpret_cast<const uint4*>(pb[1]);
    uint4 sb2 = *reinterpret_cast<const uint4*>(pb[2]);
    uint4 sb3 = *reinterpret_cast<const uint4*>(pb[3]);
    *reinterpret_cast<uint4*>(&Alds[0][lA0])   = sa0;
    *reinterpret_cast<uint4*>(&Alds[0][lA1])   = sa1;
    *reinterpret_cast<uint4*>(&Blds[0][lB[0]]) = sb0;
    *reinterpret_cast<uint4*>(&Blds[0][lB[1]]) = sb1;
    *reinterpret_cast<uint4*>(&Blds[0][lB[2]]) = sb2;
    *reinterpret_cast<uint4*>(&Blds[0][lB[3]]) = sb3;
  }
  __syncthreads();

  union Frag { v16bf v; uint4 q[2]; };
  const int arbase = (wm * 16 + lr) * GEMM_KS;             // + s*32 + half*8 (+16)
  const int brbase = (wn * 64 + lr) * GEMM_KS + half * 8;  // + s*32 + i*16*KS (+16)

  int cur = 0;
  for (int k0 = 0; k0 < K; k0 += 64) {
    const bool has_next = (k0 + 64 < K);
    const unsigned short* Ab = Alds[cur];
    const unsigned short* Bb = Blds[cur];

    // ---- sub-slab 0: 10 ds_loads -> 4 WMMAs ----
    Frag af0;
    af0.q[0] = *reinterpret_cast<const uint4*>(&Ab[arbase + half * 8]);
    af0.q[1] = *reinterpret_cast<const uint4*>(&Ab[arbase + 16 + half * 8]);
    Frag bf0[4];
#pragma unroll
    for (int i = 0; i < 4; ++i) {
      const int rb0 = brbase + i * 16 * GEMM_KS;
      bf0[i].q[0] = *reinterpret_cast<const uint4*>(&Bb[rb0]);
      bf0[i].q[1] = *reinterpret_cast<const uint4*>(&Bb[rb0 + 16]);
    }
    __builtin_amdgcn_sched_barrier(0);
    acc[0] = __builtin_amdgcn_wmma_f32_16x16x32_bf16(false, af0.v, false, bf0[0].v,
                                                     (short)0, acc[0], false, false);
    acc[1] = __builtin_amdgcn_wmma_f32_16x16x32_bf16(false, af0.v, false, bf0[1].v,
                                                     (short)0, acc[1], false, false);
    acc[2] = __builtin_amdgcn_wmma_f32_16x16x32_bf16(false, af0.v, false, bf0[2].v,
                                                     (short)0, acc[2], false, false);
    acc[3] = __builtin_amdgcn_wmma_f32_16x16x32_bf16(false, af0.v, false, bf0[3].v,
                                                     (short)0, acc[3], false, false);
    __builtin_amdgcn_sched_barrier(0);

    // ---- stage next slab (part 1): 4 chunks, live only across sub-slab 1 ----
    uint4 sa0, sa1, sb0, sb1;
    if (has_next) {
      sa0 = *reinterpret_cast<const uint4*>(pa0 + 64);
      sa1 = *reinterpret_cast<const uint4*>(pa1 + 64);
      sb0 = *reinterpret_cast<const uint4*>(pb[0] + 64);
      sb1 = *reinterpret_cast<const uint4*>(pb[1] + 64);
      if (k0 + 128 < K) {  // pull slab k0+128 into GL2 -> global_prefetch_b8
        __builtin_prefetch(pa0 + 128, 0, 0);
        __builtin_prefetch(pb[0] + 128, 0, 0);
      }
    }

    // ---- sub-slab 1: 10 ds_loads -> 4 WMMAs ----
    Frag af1;
    af1.q[0] = *reinterpret_cast<const uint4*>(&Ab[arbase + 32 + half * 8]);
    af1.q[1] = *reinterpret_cast<const uint4*>(&Ab[arbase + 48 + half * 8]);
    Frag bf1[4];
#pragma unroll
    for (int i = 0; i < 4; ++i) {
      const int rb0 = brbase + i * 16 * GEMM_KS + 32;
      bf1[i].q[0] = *reinterpret_cast<const uint4*>(&Bb[rb0]);
      bf1[i].q[1] = *reinterpret_cast<const uint4*>(&Bb[rb0 + 16]);
    }
    __builtin_amdgcn_sched_barrier(0);
    acc[0] = __builtin_amdgcn_wmma_f32_16x16x32_bf16(false, af1.v, false, bf1[0].v,
                                                     (short)0, acc[0], false, false);
    acc[1] = __builtin_amdgcn_wmma_f32_16x16x32_bf16(false, af1.v, false, bf1[1].v,
                                                     (short)0, acc[1], false, false);
    acc[2] = __builtin_amdgcn_wmma_f32_16x16x32_bf16(false, af1.v, false, bf1[2].v,
                                                     (short)0, acc[2], false, false);
    acc[3] = __builtin_amdgcn_wmma_f32_16x16x32_bf16(false, af1.v, false, bf1[3].v,
                                                     (short)0, acc[3], false, false);
    __builtin_amdgcn_sched_barrier(0);

    if (has_next) {
      // ---- stage next slab (part 2): fragments are dead here, pressure is low ----
      uint4 sb2 = *reinterpret_cast<const uint4*>(pb[2] + 64);
      uint4 sb3 = *reinterpret_cast<const uint4*>(pb[3] + 64);
      const int nxt = cur ^ 1;
      // Writes target the opposite buffer from all in-flight reads -> one barrier.
      *reinterpret_cast<uint4*>(&Alds[nxt][lA0])   = sa0;
      *reinterpret_cast<uint4*>(&Alds[nxt][lA1])   = sa1;
      *reinterpret_cast<uint4*>(&Blds[nxt][lB[0]]) = sb0;
      *reinterpret_cast<uint4*>(&Blds[nxt][lB[1]]) = sb1;
      *reinterpret_cast<uint4*>(&Blds[nxt][lB[2]]) = sb2;
      *reinterpret_cast<uint4*>(&Blds[nxt][lB[3]]) = sb3;
      __syncthreads();
      cur = nxt;
      pa0 += 64; pa1 += 64;
#pragma unroll
      for (int j = 0; j < 4; ++j) pb[j] += 64;
    }
  }

  // Epilogue. C/D layout: VGPR r2 -> M = r2 + half*8, N = lr (within 16x16 subtile).
#pragma unroll
  for (int i = 0; i < 4; ++i) {
    const int col = bn + wn * 64 + i * 16 + lr;
    const float sc = scale ? scale[col] : 1.0f;
#pragma unroll
    for (int r2 = 0; r2 < 8; ++r2) {
      const int row = bm + wm * 16 + half * 8 + r2;
      float v = acc[i][r2] * sc;
      if (mode == 1)      v = sigmoidf_(v);
      else if (mode == 2) v = v * sigmoidf_(v);
      const size_t off = (size_t)row * N + col;
      if (mulp) v *= mulp[off];
      if (addp) v += addp[off];
      C[off] = v;
    }
  }
}

// ---------------------------------------------------------------------------
extern "C" void kernel_launch(void* const* d_in, const int* in_sizes, int n_in,
                              void* d_out, int out_size, void* d_ws, size_t ws_size,
                              hipStream_t stream) {
  (void)in_sizes; (void)n_in; (void)out_size; (void)ws_size;
  constexpr int V = 32000, D = 1024, L = 6, FF = 4096, Bb = 2, T = 1024;
  constexpr int M = Bb * T;  // 2048 tokens

  const int*   idx      = (const int*)d_in[0];
  const float* embed    = (const float*)d_in[1];
  const float* ln_in_w  = (const float*)d_in[2];
  const float* ln1_w    = (const float*)d_in[3];
  const float* Wr       = (const float*)d_in[4];
  const float* Wk       = (const float*)d_in[5];
  const float* Wv       = (const float*)d_in[6];
  const float* Wo_t     = (const float*)d_in[7];
  const float* decay    = (const float*)d_in[8];
  const float* lnx_w    = (const float*)d_in[9];
  const float* ln2_w    = (const float*)d_in[10];
  const float* W1       = (const float*)d_in[11];
  const float* W2       = (const float*)d_in[12];
  const float* Wo_c     = (const float*)d_in[13];
  const float* ln_out_w = (const float*)d_in[14];
  float* out = (float*)d_out;

  // ---- workspace carve (all 256B aligned) ----
  char* p = (char*)d_ws;
  auto carve = [&](size_t bytes) -> char* {
    char* r = p; p += (bytes + 255) & ~((size_t)255); return r;
  };
  float* h    = (float*)carve((size_t)M * D * 4);
  float* rb   = (float*)carve((size_t)M * D * 4);
  float* kb   = (float*)carve((size_t)M * D * 4);
  float* vb   = (float*)carve((size_t)M * D * 4);
  float* rs   = (float*)carve((size_t)M * D * 4);
  float* ff1  = (float*)carve((size_t)M * FF * 4);
  float* ff2  = (float*)carve((size_t)M * FF * 4);
  unsigned short* xnb = (unsigned short*)carve((size_t)M * D * 2);
  unsigned short* ffb = (unsigned short*)carve((size_t)M * FF * 2);
  unsigned short* qW  = (unsigned short*)carve(((size_t)4 * D * D + 3 * (size_t)FF * D) * 2);
  float* sW           = (float*)carve(((size_t)4 * D + 2 * FF + D) * 4);
  unsigned short* ebf = (unsigned short*)carve((size_t)V * D * 2);

  unsigned short* qWr  = qW;
  unsigned short* qWk  = qW + (size_t)1 * D * D;
  unsigned short* qWv  = qW + (size_t)2 * D * D;
  unsigned short* qWo  = qW + (size_t)3 * D * D;
  unsigned short* qW1  = qW + (size_t)4 * D * D;
  unsigned short* qW2  = qW + (size_t)4 * D * D + (size_t)FF * D;
  unsigned short* qWoc = qW + (size_t)4 * D * D + (size_t)2 * FF * D;
  float* sr  = sW;
  float* sk  = sW + D;
  float* sv  = sW + 2 * D;
  float* so_ = sW + 3 * D;
  float* s1  = sW + 4 * D;
  float* s2  = sW + 4 * D + FF;
  float* soc = sW + 4 * D + 2 * FF;

  // embed -> bf16 (B operand of the weight-tied head)
  cast_bf16_kernel<<<2048, 256, 0, stream>>>(embed, ebf, (long long)V * D);
  // h = rmsnorm(embed[idx], ln_in_w)
  gather_rmsnorm_kernel<<<M, 256, 0, stream>>>(idx, embed, ln_in_w, h, D);

  const dim3 gDD(D / 128, M / 64);    // N=1024 GEMMs
  const dim3 gDF(FF / 128, M / 64);   // N=4096 GEMMs

  for (int i = 0; i < L; ++i) {
    // --- ternary-quantize this layer's weights (bf16 + per-row scale) ---
    quantize_rows_kernel<<<D,  256, 0, stream>>>(Wr   + (size_t)i * D * D,  qWr,  sr,  D);
    quantize_rows_kernel<<<D,  256, 0, stream>>>(Wk   + (size_t)i * D * D,  qWk,  sk,  D);
    quantize_rows_kernel<<<D,  256, 0, stream>>>(Wv   + (size_t)i * D * D,  qWv,  sv,  D);
    quantize_rows_kernel<<<D,  256, 0, stream>>>(Wo_t + (size_t)i * D * D,  qWo,  so_, D);
    quantize_rows_kernel<<<FF, 256, 0, stream>>>(W1   + (size_t)i * FF * D, qW1,  s1,  D);
    quantize_rows_kernel<<<FF, 256, 0, stream>>>(W2   + (size_t)i * FF * D, qW2,  s2,  D);
    quantize_rows_kernel<<<D,  256, 0, stream>>>(Wo_c + (size_t)i * D * FF, qWoc, soc, FF);

    // --- time mix ---
    rmsnorm_cast_kernel<<<M, 256, 0, stream>>>(h, ln1_w + (size_t)i * D, xnb, D);
    gemm_bf16_kernel<<<gDD, 256, 0, stream>>>(xnb, qWr, sr, nullptr, nullptr, rb, M, D, D, 1); // r = sigmoid
    gemm_bf16_kernel<<<gDD, 256, 0, stream>>>(xnb, qWk, sk, nullptr, nullptr, kb, M, D, D, 0);
    gemm_bf16_kernel<<<gDD, 256, 0, stream>>>(xnb, qWv, sv, nullptr, nullptr, vb, M, D, D, 0);
    wkv_scan_kernel<<<(Bb * D + 255) / 256, 256, 0, stream>>>(rb, kb, vb, decay + (size_t)i * D,
                                                              rs, Bb, T, D);
    rmsnorm_cast_kernel<<<M, 256, 0, stream>>>(rs, lnx_w + (size_t)i * D, xnb, D);
    gemm_bf16_kernel<<<gDD, 256, 0, stream>>>(xnb, qWo, so_, h, nullptr, h, M, D, D, 0); // h += proj

    // --- channel mix ---
    rmsnorm_cast_kernel<<<M, 256, 0, stream>>>(h, ln2_w + (size_t)i * D, xnb, D);
    gemm_bf16_kernel<<<gDF, 256, 0, stream>>>(xnb, qW1, s1, nullptr, nullptr, ff1, M, FF, D, 2); // silu
    gemm_bf16_kernel<<<gDF, 256, 0, stream>>>(xnb, qW2, s2, nullptr, ff1,     ff2, M, FF, D, 0); // * gate
    cast_bf16_kernel<<<2048, 256, 0, stream>>>(ff2, ffb, (long long)M * FF);
    gemm_bf16_kernel<<<gDD, 256, 0, stream>>>(ffb, qWoc, soc, h, nullptr, h, M, D, FF, 0); // h += proj
  }

  // --- weight-tied head: out = rmsnorm(h, ln_out_w) @ embed^T ---
  rmsnorm_cast_kernel<<<M, 256, 0, stream>>>(h, ln_out_w, xnb, D);
  const dim3 gHead(V / 128, M / 64);
  gemm_bf16_kernel<<<gHead, 256, 0, stream>>>(xnb, ebf, nullptr, nullptr, nullptr, out, M, V, D, 0);
}
